// DeltaNet_31877247271570
// MI455X (gfx1250) — matrix operand order repaired and checked by reading, not compiled, not run on hardware
//
#include <hip/hip_runtime.h>
#include <hip/hip_bf16.h>
#include <cstdint>

// ---------------- problem constants ----------------
static constexpr int B_   = 4;
static constexpr int H_   = 8;
static constexpr int LSEQ = 8192;
static constexpr int DK   = 64;
static constexpr int DV   = 64;
static constexpr int CHK  = 32;
static constexpr int NCH  = LSEQ / CHK;                    // 256 chunks
static constexpr long NTOT = (long)B_ * H_ * LSEQ * DK;    // 16,777,216 elems

typedef __attribute__((ext_vector_type(16))) __bf16 v16bf;
typedef __attribute__((ext_vector_type(8)))  float  v8f;

// ---------------- WMMA fragment helpers (wave32, ISA 7.12.2 layouts) ----------------

// A fragment (16x32 bf16 tile, M x K), source row-major, leading dim `ld`.
// lane L: M = row0 + (L&15); half = L>>4
// element e: v=e>>1, s=e&1 ; K = 16*(v>>2) + 8*half + 2*(v&3) + s
__device__ __forceinline__ v16bf frag_a(const __bf16* src, int ld, int row0, int k0) {
  int lane = threadIdx.x & 31;
  int m    = row0 + (lane & 15);
  int half = lane >> 4;
  v16bf a;
#pragma unroll
  for (int e = 0; e < 16; ++e) {
    int v = e >> 1, s = e & 1;
    int k = k0 + ((v >> 2) << 4) + (half << 3) + ((v & 3) << 1) + s;
    a[e] = src[m * ld + k];
  }
  return a;
}

// A fragment where the SOURCE stores A transposed: A(M,K) = src[(k0+K)*ld + row0+M]
__device__ __forceinline__ v16bf frag_aT(const __bf16* src, int ld, int row0, int k0) {
  int lane = threadIdx.x & 31;
  int m    = row0 + (lane & 15);
  int half = lane >> 4;
  v16bf a;
#pragma unroll
  for (int e = 0; e < 16; ++e) {
    int v = e >> 1, s = e & 1;
    int k = k0 + ((v >> 2) << 4) + (half << 3) + ((v & 3) << 1) + s;
    a[e] = src[k * ld + m];
  }
  return a;
}

// B fragment (32x16 bf16 tile, K x N), SOURCE stored transposed: srcT[n][k] == B(k,n)
// lane L: N = col0 + (L&15); element e: K = 16*(e>>3) + 8*(L>>4) + (e&7)
__device__ __forceinline__ v16bf frag_bT(const __bf16* srcT, int ld, int k0, int col0) {
  int lane = threadIdx.x & 31;
  int n    = col0 + (lane & 15);
  int kw   = (lane >> 4) << 3;
  v16bf b;
#pragma unroll
  for (int e = 0; e < 16; ++e) {
    b[e] = srcT[n * ld + k0 + kw + ((e >> 3) << 4) + (e & 7)];
  }
  return b;
}

// C/D fragment (16x16 f32): lane L: N = col0 + (L&15); rows M = row0 + 8*(L>>4) + r
__device__ __forceinline__ v8f frag_c_load(const float* src, int ld, int row0, int col0) {
  int lane = threadIdx.x & 31;
  int n  = col0 + (lane & 15);
  int m0 = row0 + ((lane >> 4) << 3);
  v8f c;
#pragma unroll
  for (int r = 0; r < 8; ++r) c[r] = src[(m0 + r) * ld + n];
  return c;
}

__device__ __forceinline__ void frag_c_store(float* dst, int ld, int row0, int col0, v8f c) {
  int lane = threadIdx.x & 31;
  int n  = col0 + (lane & 15);
  int m0 = row0 + ((lane >> 4) << 3);
#pragma unroll
  for (int r = 0; r < 8; ++r) dst[(m0 + r) * ld + n] = c[r];
}

__device__ __forceinline__ v8f wmma_bf16(v16bf a, v16bf b, v8f c) {
  return __builtin_amdgcn_wmma_f32_16x16x32_bf16(false, a, false, b, (short)0, c, false, false);
}

// ---------------- kernel 1: chunk-parallel preprocessing ----------------
__global__ void dn_pre(const float* __restrict__ q, const float* __restrict__ k,
                       const float* __restrict__ v, const float* __restrict__ beta,
                       __bf16* __restrict__ qn, __bf16* __restrict__ kn,
                       __bf16* __restrict__ u0, __bf16* __restrict__ w) {
  __shared__ float qs[32 * 64], ks[32 * 64], vs[32 * 64], kbs[32 * 64];
  __shared__ float att[32 * 32];
  __shared__ float bet[32];

  const int tid = threadIdx.x;
  const long base = (long)blockIdx.x * (CHK * DK);

  for (int i = tid; i < 2048; i += 256) {
    qs[i] = q[base + i]; ks[i] = k[base + i]; vs[i] = v[base + i];
  }
  if (tid < 32) bet[tid] = beta[(long)blockIdx.x * CHK + tid];
  __syncthreads();

  if (tid < 32) {                               // per-row l2norm + beta scale
    const int r = tid;
    float sq = 1e-6f, sk = 1e-6f;
    for (int d = 0; d < 64; ++d) {
      float a = qs[r * 64 + d]; sq += a * a;
      float b = ks[r * 64 + d]; sk += b * b;
    }
    const float rq = rsqrtf(sq), rk = rsqrtf(sk), bb = bet[r];
    for (int d = 0; d < 64; ++d) {
      qs[r * 64 + d] *= rq;
      float kv = ks[r * 64 + d] * rk;
      ks[r * 64 + d]  = kv;
      kbs[r * 64 + d] = kv * bb;
      vs[r * 64 + d] *= bb;
    }
  }
  __syncthreads();

  // attn = -(k_beta @ k^T), strictly lower triangular
  for (int e = tid; e < 1024; e += 256) {
    int i = e >> 5, j = e & 31;
    float s = 0.f;
    if (j < i) {
      for (int d = 0; d < 64; ++d) s -= kbs[i * 64 + d] * ks[j * 64 + d];
    }
    att[e] = s;
  }
  __syncthreads();

  // forward substitution -> (I - A)^{-1} strict-lower part
  for (int i = 1; i < 32; ++i) {
    float val = 0.f;
    const int m = tid;
    if (m < i) {
      float s = 0.f;
      for (int j = m + 1; j < i; ++j) s += att[i * 32 + j] * att[j * 32 + m];
      val = att[i * 32 + m] + s;
    }
    __syncthreads();
    if (m < i) att[i * 32 + m] = val;
    __syncthreads();
  }
  if (tid < 32) att[tid * 32 + tid] = 1.0f;     // + I
  __syncthreads();

  // u = T @ v_beta ; w = T @ k_beta ; emit bf16 (plus normalized q,k)
  for (int e = tid; e < 2048; e += 256) {
    int i = e >> 6, d = e & 63;
    float su = 0.f, sw = 0.f;
    for (int j = 0; j <= i; ++j) {
      float a = att[i * 32 + j];
      su += a * vs[j * 64 + d];
      sw += a * kbs[j * 64 + d];
    }
    u0[base + e] = (__bf16)su;
    w [base + e] = (__bf16)sw;
    qn[base + e] = (__bf16)qs[e];
    kn[base + e] = (__bf16)ks[e];
  }
}

// ---------------- kernel 2: sequential inter-chunk scan (WMMA + async staging) ----------------
// One block (8 wave32s) per (b,h).  Chunk operands are double-buffered in LDS
// via GLOBAL_LOAD_ASYNC_TO_LDS_B128 (ASYNCcnt) so global latency overlaps the
// WMMA work of the current chunk.  S kept f32 in LDS.
__global__ void dn_scan(const __bf16* __restrict__ qn, const __bf16* __restrict__ kn,
                        const __bf16* __restrict__ u0g, const __bf16* __restrict__ wg,
                        float* __restrict__ out) {
  __shared__ __align__(16) __bf16 stage[2][4][2048];  // {q,k,u0,w} x double buffer, 32 KB
  __shared__ float  S  [64 * 64];                     // running state f32, 16 KB
  __shared__ __bf16 STb[64 * 72];                     // S^T bf16, padded rows
  __shared__ __bf16 uTb[64 * 40];                     // u^T bf16, padded rows
  __shared__ __bf16 attb[32 * 32];

  const int tid  = threadIdx.x;
  const int lane = tid & 31;
  const int wv   = tid >> 5;
  const int bh   = blockIdx.x;

  const __bf16* srcs[4] = { qn, kn, u0g, wg };

  // issue this wave's 4 async 512B slices for chunk c into buffer pb
  auto issue = [&](int c, int pb) {
    const long base = ((long)bh * NCH + c) * (CHK * DK);
#pragma unroll
    for (int t = 0; t < 4; ++t) {
      const int s   = (wv << 2) + t;                  // slice 0..31
      const int buf = s >> 3;
      const int off = ((s & 7) << 8) + (lane << 3);   // 256 elems/slice, 8/lane
      unsigned    laddr = (unsigned)(uintptr_t)(&stage[pb][buf][off]);
      const void* g     = (const void*)(srcs[buf] + base + off);
      asm volatile("global_load_async_to_lds_b128 %0, %1, off"
                   :: "v"(laddr), "v"(g) : "memory");
    }
  };

  issue(0, 0);                                        // prefetch chunk 0
  for (int i = tid; i < 4096; i += 256) S[i] = 0.f;
  __syncthreads();

  const long obase0 = (long)bh * LSEQ * DV;

  for (int c = 0; c < NCH; ++c) {
    const int pb = c & 1;

    if (c + 1 < NCH) {
      issue(c + 1, pb ^ 1);                           // prefetch next chunk
      asm volatile("s_wait_asynccnt 4" ::: "memory"); // chunk c's batch done
    } else {
      asm volatile("s_wait_asynccnt 0" ::: "memory");
    }
    __syncthreads();                                  // staging visible to all waves

    const __bf16* qb  = stage[pb][0];
    const __bf16* kbm = stage[pb][1];
    const __bf16* u0b = stage[pb][2];
    const __bf16* wb  = stage[pb][3];

    // ---- waves 0..3: attn_local = tril(q @ k^T) | waves 4..7: rebuild S^T bf16 ----
    if (wv < 4) {
      const int ti = (wv >> 1) << 4, tj = (wv & 1) << 4;
      v8f acc = {};
      acc = wmma_bf16(frag_a(qb, 64, ti, 0),  frag_bT(kbm, 64, 0,  tj), acc);
      acc = wmma_bf16(frag_a(qb, 64, ti, 32), frag_bT(kbm, 64, 32, tj), acc);
      const int n  = tj + (lane & 15);
      const int m0 = ti + ((lane >> 4) << 3);
#pragma unroll
      for (int r = 0; r < 8; ++r) {
        float vvv = ((m0 + r) >= n) ? acc[r] : 0.f;   // causal incl. diagonal
        attb[(m0 + r) * 32 + n] = (__bf16)vvv;
      }
    } else {
      for (int i = tid - 128; i < 4096; i += 128) {
        int d1 = i >> 6, d2 = i & 63;
        STb[d2 * 72 + d1] = (__bf16)S[d1 * 64 + d2];
      }
    }
    __syncthreads();

    // ---- u_i = u0 - w @ S : 8 tiles, one per wave; write u^T bf16 ----
    {
      const int ti = (wv >> 2) << 4, tj = (wv & 3) << 4;
      v8f acc = {};
      acc = wmma_bf16(frag_a(wb, 64, ti, 0),  frag_bT(STb, 72, 0,  tj), acc);
      acc = wmma_bf16(frag_a(wb, 64, ti, 32), frag_bT(STb, 72, 32, tj), acc);
      const int n  = tj + (lane & 15);
      const int m0 = ti + ((lane >> 4) << 3);
#pragma unroll
      for (int r = 0; r < 8; ++r) {
        float uu = (float)u0b[(m0 + r) * 64 + n] - acc[r];
        uTb[n * 40 + (m0 + r)] = (__bf16)uu;
      }
    }
    __syncthreads();

    // ---- o_i = q @ S + attn_local @ u_i : 8 tiles, one per wave ----
    {
      const int ti = (wv >> 2) << 4, tj = (wv & 3) << 4;
      v8f acc = {};
      acc = wmma_bf16(frag_a(qb, 64, ti, 0),   frag_bT(STb, 72, 0,  tj), acc);
      acc = wmma_bf16(frag_a(qb, 64, ti, 32),  frag_bT(STb, 72, 32, tj), acc);
      acc = wmma_bf16(frag_a(attb, 32, ti, 0), frag_bT(uTb, 40, 0,  tj), acc);
      const int n  = tj + (lane & 15);
      const int m0 = ti + ((lane >> 4) << 3);
      const long ob = obase0 + (long)c * CHK * DV;
#pragma unroll
      for (int r = 0; r < 8; ++r) out[ob + (long)(m0 + r) * DV + n] = acc[r];
    }

    // ---- S += k^T @ u_i : 16 tiles, two per wave (C operand = S tile) ----
#pragma unroll
    for (int t = 0; t < 2; ++t) {
      const int tile = wv + (t << 3);
      const int ti = (tile >> 2) << 4, tj = (tile & 3) << 4;
      v8f cfr = frag_c_load(S, 64, ti, tj);
      cfr = wmma_bf16(frag_aT(kbm, 64, ti, 0), frag_bT(uTb, 40, 0, tj), cfr);
      frag_c_store(S, 64, ti, tj, cfr);
    }
    __syncthreads();
  }

  // final state S -> tail of d_out
  float* Sout = out + (long)B_ * H_ * LSEQ * DV + (long)bh * DK * DV;
  for (int i = tid; i < 4096; i += 256) Sout[i] = S[i];
}

// ---------------- host launcher ----------------
extern "C" void kernel_launch(void* const* d_in, const int* in_sizes, int n_in,
                              void* d_out, int out_size, void* d_ws, size_t ws_size,
                              hipStream_t stream) {
  const float* q    = (const float*)d_in[0];
  const float* k    = (const float*)d_in[1];
  const float* v    = (const float*)d_in[2];
  const float* beta = (const float*)d_in[3];

  __bf16* ws = (__bf16*)d_ws;            // 4 * NTOT bf16 = 128 MB
  __bf16* qn = ws;
  __bf16* kn = ws + NTOT;
  __bf16* u0 = ws + 2 * NTOT;
  __bf16* w  = ws + 3 * NTOT;

  float* out = (float*)d_out;

  dn_pre <<<B_ * H_ * NCH, 256, 0, stream>>>(q, k, v, beta, qn, kn, u0, w);
  dn_scan<<<B_ * H_,       256, 0, stream>>>(qn, kn, u0, w, out);
}
